// COINBlock_51608327029364
// MI455X (gfx1250) — matrix operand order repaired and checked by reading, not compiled, not run on hardware
//
#include <hip/hip_runtime.h>
#include <hip/hip_bf16.h>

typedef __attribute__((ext_vector_type(16))) __bf16 bf16x16;
typedef __attribute__((ext_vector_type(8)))  __bf16 bf16x8;
typedef __attribute__((ext_vector_type(8)))  float  f32x8;
typedef __attribute__((ext_vector_type(4)))  float  f32x4;

#define T_DIM 4096
#define K_DIM 2048
#define N_DIM 2048
#define BM 128
#define BN 64
#define BK 32
#define LDA 40   // padded LDS row stride in bf16 (80B: conflict-free b128 rows)
#define LDB 40
#define KSTEPS (K_DIM / BK)

union AFrag { bf16x16 v; bf16x8 h[2]; };

__global__ __launch_bounds__(256)
void fused_qrope_gn_kernel(const float* __restrict__ X,
                           const float* __restrict__ W,
                           const float* __restrict__ hh_b,
                           const float* __restrict__ gn_w,
                           const float* __restrict__ gn_b,
                           float* __restrict__ out)
{
    __shared__ alignas(16) __bf16 AsH[BM * LDA];
    __shared__ alignas(16) __bf16 AsL[BM * LDA];
    __shared__ alignas(16) __bf16 BsH[BN * LDB];
    __shared__ alignas(16) __bf16 BsL[BN * LDB];

    const int tid     = threadIdx.x;
    const int wave    = tid >> 5;
    const int lane    = tid & 31;
    const int rsel    = lane & 15;   // row (A) / col (B) selector within 16
    const int kh      = lane >> 4;   // K-half selector
    const int rowBase = blockIdx.y * BM;   // global row m in [0, 8192)
    const int colBase = blockIdx.x * BN;   // channel block (one GN group)

    f32x8 acc[4];
#pragma unroll
    for (int j = 0; j < 4; ++j)
#pragma unroll
        for (int e = 0; e < 8; ++e) acc[j][e] = 0.0f;

    for (int kb = 0; kb < KSTEPS; ++kb) {
        // ---- stage A tile (128x32 f32) as bf16 hi/lo, row-major ----
#pragma unroll
        for (int p = 0; p < 4; ++p) {
            int g   = tid + p * 256;          // 1024 float4 chunks
            int row = g >> 3;                 // 8 float4 per 32-wide row
            int c4  = g & 7;
            const float4 v = *(const float4*)(X + (size_t)(rowBase + row) * K_DIM
                                                + kb * BK + c4 * 4);
            const float vv[4] = {v.x, v.y, v.z, v.w};
#pragma unroll
            for (int i = 0; i < 4; ++i) {
                __bf16 hi = (__bf16)vv[i];
                __bf16 lo = (__bf16)(vv[i] - (float)hi);
                AsH[row * LDA + c4 * 4 + i] = hi;
                AsL[row * LDA + c4 * 4 + i] = lo;
            }
        }
        // ---- stage B tile (32x64 f32) transposed -> [n][k] bf16 hi/lo ----
#pragma unroll
        for (int p = 0; p < 2; ++p) {
            int g  = tid + p * 256;           // 512 float4 chunks
            int k  = g >> 4;                  // 16 float4 per 64-wide row
            int c4 = g & 15;
            const float4 v = *(const float4*)(W + (size_t)(kb * BK + k) * N_DIM
                                                + colBase + c4 * 4);
            const float vv[4] = {v.x, v.y, v.z, v.w};
#pragma unroll
            for (int i = 0; i < 4; ++i) {
                __bf16 hi = (__bf16)vv[i];
                __bf16 lo = (__bf16)(vv[i] - (float)hi);
                BsH[(c4 * 4 + i) * LDB + k] = hi;
                BsL[(c4 * 4 + i) * LDB + k] = lo;
            }
        }
        // prefetch next K-slab (global_prefetch_b8)
        if (kb + 1 < KSTEPS) {
            __builtin_prefetch(X + (size_t)(rowBase + (tid >> 3)) * K_DIM
                                 + (kb + 1) * BK + (tid & 7) * 4, 0, 1);
            __builtin_prefetch(W + (size_t)((kb + 1) * BK + (tid >> 4)) * N_DIM
                                 + colBase + (tid & 15) * 4, 0, 1);
        }
        __syncthreads();

        // ---- A fragments for this wave's 16 rows ----
        AFrag aH, aL;
        {
            const __bf16* arH = &AsH[(16 * wave + rsel) * LDA];
            const __bf16* arL = &AsL[(16 * wave + rsel) * LDA];
            aH.h[0] = *(const bf16x8*)(arH + 8 * kh);
            aH.h[1] = *(const bf16x8*)(arH + 16 + 8 * kh);
            aL.h[0] = *(const bf16x8*)(arL + 8 * kh);
            aL.h[1] = *(const bf16x8*)(arL + 16 + 8 * kh);
        }
#pragma unroll
        for (int j = 0; j < 4; ++j) {
            AFrag bH, bL;
            const __bf16* brH = &BsH[(16 * j + rsel) * LDB + 8 * kh];
            const __bf16* brL = &BsL[(16 * j + rsel) * LDB + 8 * kh];
            bH.h[0] = *(const bf16x8*)(brH);
            bH.h[1] = *(const bf16x8*)(brH + 16);
            bL.h[0] = *(const bf16x8*)(brL);
            bL.h[1] = *(const bf16x8*)(brL + 16);
            // bf16x3: AhiBhi + AloBhi + AhiBlo, fp32 accumulate
            acc[j] = __builtin_amdgcn_wmma_f32_16x16x32_bf16(
                false, aH.v, false, bH.v, (short)0, acc[j], false, false);
            acc[j] = __builtin_amdgcn_wmma_f32_16x16x32_bf16(
                false, aL.v, false, bH.v, (short)0, acc[j], false, false);
            acc[j] = __builtin_amdgcn_wmma_f32_16x16x32_bf16(
                false, aH.v, false, bL.v, (short)0, acc[j], false, false);
        }
        __syncthreads();
    }

    // ================= fused epilogue =================
    // Lane holds: channel c_j = colBase + 16j + rsel (fixed per lane),
    // rows m = rowBase + 16*wave + 8*kh + r, r = 0..7.
    const int m0 = rowBase + 16 * wave + 8 * kh;
    const int b  = m0 / T_DIM;
    const int t0 = m0 % T_DIM;           // 8 consecutive t values per lane

    float hb[4], gw[4], gb[4];
#pragma unroll
    for (int j = 0; j < 4; ++j) {
        int c = colBase + 16 * j + rsel;
        hb[j] = hh_b[c];
        gw[j] = gn_w[c];
        gb[j] = gn_b[c];
    }

    const bool  doRope = (colBase == 0);          // rope hits channels 0..63 only
    const float sgn    = (lane & 1) ? 1.0f : -1.0f;
    float res[4][8];

#pragma unroll
    for (int r = 0; r < 8; ++r) {
        const int t = t0 + r;
        float s = 0.0f, sq = 0.0f;
        float hv[4];
#pragma unroll
        for (int j = 0; j < 4; ++j) {
            float y = acc[j][r];
            if (doRope) {
                // pair partner lives in lane^1 (same channel pair 2i,2i+1)
                float other = __shfl_xor(y, 1, 32);
                int   i     = (16 * j + rsel) >> 1;           // pair index 0..31
                float theta = (float)t * __powf(10000.0f, -(float)i * (1.0f / 32.0f));
                float sn, cs;
                __sincosf(theta, &sn, &cs);
                y = y * cs + sgn * other * sn;
            }
            y = fmaxf(y, 0.0f);                                // relu
            float h = tanhf(y + hb[j]);
            hv[j] = h;
            s  += h;
            sq += h * h;
        }
        // GroupNorm stats over the 64 channels of this row (16 lanes x 4 j);
        // xor masks 1,2,4,8 keep the two lane-halves (two rows) independent.
#pragma unroll
        for (int mask = 1; mask <= 8; mask <<= 1) {
            s  += __shfl_xor(s,  mask, 32);
            sq += __shfl_xor(sq, mask, 32);
        }
        const float mu  = s * (1.0f / 64.0f);
        const float var = sq * (1.0f / 64.0f) - mu * mu;
        const float inv = rsqrtf(var + 1e-5f);
#pragma unroll
        for (int j = 0; j < 4; ++j)
            res[j][r] = (hv[j] - mu) * inv * gw[j] + gb[j];
    }

    // transposed store: flat index b*T*C + c*T + t  (8 consecutive t per lane)
#pragma unroll
    for (int j = 0; j < 4; ++j) {
        const size_t c  = (size_t)(colBase + 16 * j + rsel);
        float* po = out + (size_t)b * T_DIM * N_DIM + c * T_DIM + t0;
        f32x4 lo4 = {res[j][0], res[j][1], res[j][2], res[j][3]};
        f32x4 hi4 = {res[j][4], res[j][5], res[j][6], res[j][7]};
        __builtin_nontemporal_store(lo4, (f32x4*)po);
        __builtin_nontemporal_store(hi4, (f32x4*)(po + 4));
    }
}

__global__ void copy_sn_kernel(const float* __restrict__ s,
                               float* __restrict__ dst, int n)
{
    int i = blockIdx.x * blockDim.x + threadIdx.x;
    if (i < n) dst[i] = s[i];
}

extern "C" void kernel_launch(void* const* d_in, const int* in_sizes, int n_in,
                              void* d_out, int out_size, void* d_ws, size_t ws_size,
                              hipStream_t stream)
{
    const float* X    = (const float*)d_in[0];
    const float* S_n  = (const float*)d_in[1];
    const float* W_Q  = (const float*)d_in[2];
    const float* hh_b = (const float*)d_in[3];
    const float* gn_w = (const float*)d_in[4];
    const float* gn_b = (const float*)d_in[5];
    float* out = (float*)d_out;

    dim3 grid(N_DIM / BN, (2 * T_DIM * 2) / BM / 2);   // (32, 64): M = B*T = 8192
    fused_qrope_gn_kernel<<<grid, 256, 0, stream>>>(X, W_Q, hh_b, gn_w, gn_b, out);

    const int snN = 2 * N_DIM;                          // B*C = 4096
    copy_sn_kernel<<<(snN + 255) / 256, 256, 0, stream>>>(
        S_n, out + (size_t)2 * T_DIM * N_DIM, snN);
}
